// PolicyLoss_12017318494815
// MI455X (gfx1250) — matrix operand order repaired and checked by reading, not compiled, not run on hardware
//
#include <hip/hip_runtime.h>
#include <hip/hip_bf16.h>
#include <math.h>

typedef __attribute__((ext_vector_type(16))) _Float16 v16h;
typedef __attribute__((ext_vector_type(8)))  float    v8f;

#define NCLS 100
#define BTOT 2048
#define B1   1024
#define ODIM 9
#define MATS 16   // padded output columns per GEMM panel

// ---------------------------------------------------------------------------
// Kernel 1: the four (1024x64)@(64x9) GEMMs via v_wmma_f32_16x16x32_f16.
//  g=0: A_s = feat_s[0::2] @ W2[:, :64]^T   g=1: B_s = feat_s[1::2] @ W2[:, 64:]^T
//  g=2: A_t = feat_t[0::2] @ W1[:, :64]^T   g=3: B_t = feat_t[1::2] @ W1[:, 64:]^T
// One wave per 16-row tile; K=64 -> two WMMA steps. Outputs 1024x16 f32 panels
// (cols 9..15 zero, bias added later in the fused reduction).
// ---------------------------------------------------------------------------
__global__ void __launch_bounds__(128)
pair_linear_wmma_kernel(const float* __restrict__ s_pol,
                        const float* __restrict__ t_pol,
                        const float* __restrict__ W1,
                        const float* __restrict__ W2,
                        float* __restrict__ mats) {
  const int wid   = (blockIdx.x * blockDim.x + threadIdx.x) >> 5;  // 0..255
  const int lane  = threadIdx.x & 31;
  const int gemm  = wid >> 6;        // 0..3
  const int m0    = (wid & 63) * 16; // row tile base
  const float* feat = (gemm < 2) ? s_pol : t_pol;
  const float* W    = (gemm < 2) ? W2   : W1;
  const int half    = gemm & 1;                       // 0 -> even rows / W[:, :64]
  const int feat_row = 2 * (m0 + (lane & 15)) + half; // interleaved row select
  const int lane_hi  = lane >> 4;

  v8f acc = {};
#pragma unroll
  for (int k0 = 0; k0 < 64; k0 += 32) {
    // A fragment (16-bit A 16x32 layout): lanes<16 hold K={0..7,16..23}+k0,
    // lanes>=16 hold K={8..15,24..31}+k0, M = lane&15.
    v16h a, b;
    const int kbase = k0 + lane_hi * 8;
#pragma unroll
    for (int t = 0; t < 8; ++t) {
      a[t]     = (_Float16)feat[feat_row * 64 + kbase + t];
      a[8 + t] = (_Float16)feat[feat_row * 64 + kbase + 16 + t];
    }
    // B fragment (32x16): N = lane&15, K same as A fragment.
    // B[k][n] = W[n][64*half + k]; pad n>=9 with zeros.
    const int nb = lane & 15;
    const int kcol = half * 64 + kbase;
#pragma unroll
    for (int t = 0; t < 8; ++t) {
      b[t]     = (nb < ODIM) ? (_Float16)W[nb * 128 + kcol + t] : (_Float16)0.0f;
      b[8 + t] = (nb < ODIM) ? (_Float16)W[nb * 128 + kcol + 16 + t] : (_Float16)0.0f;
    }

    acc = __builtin_amdgcn_wmma_f32_16x16x32_f16(false, a, false, b,
                                                 (short)0, acc, false, false);
  }
  // D layout: element v -> row m0 + v + 8*(lane>>4), col lane&15.
  float* out = mats + gemm * (B1 * MATS);
#pragma unroll
  for (int v = 0; v < 8; ++v)
    out[(m0 + v + lane_hi * 8) * MATS + (lane & 15)] = acc[v];
}

// ---------------------------------------------------------------------------
// Kernel 2: CE + temperature-KL over the 2048x100 logits. One block per row.
// ---------------------------------------------------------------------------
__global__ void __launch_bounds__(128)
logits_kernel(const float* __restrict__ sl, const float* __restrict__ tl,
              const int* __restrict__ targets,
              float* __restrict__ ce_part, float* __restrict__ kl_part) {
  __shared__ float sm[128];
  const int row = blockIdx.x;
  const int tid = threadIdx.x;
  const bool ok = tid < NCLS;
  const float s = ok ? sl[row * NCLS + tid] : -1.0e30f;
  const float t = ok ? tl[row * NCLS + tid] : -1.0e30f;

  sm[tid] = s; __syncthreads();
  for (int off = 64; off; off >>= 1) { if (tid < off) sm[tid] = fmaxf(sm[tid], sm[tid + off]); __syncthreads(); }
  const float ms = sm[0]; __syncthreads();
  sm[tid] = t; __syncthreads();
  for (int off = 64; off; off >>= 1) { if (tid < off) sm[tid] = fmaxf(sm[tid], sm[tid + off]); __syncthreads(); }
  const float mt = sm[0]; __syncthreads();

  const float es  = ok ? expf(s - ms)          : 0.f;
  const float es4 = ok ? expf((s - ms) * 0.25f) : 0.f;  // 1/KL_TEMP = 0.25
  const float et4 = ok ? expf((t - mt) * 0.25f) : 0.f;

  sm[tid] = es; __syncthreads();
  for (int off = 64; off; off >>= 1) { if (tid < off) sm[tid] += sm[tid + off]; __syncthreads(); }
  const float Zs = sm[0]; __syncthreads();
  sm[tid] = es4; __syncthreads();
  for (int off = 64; off; off >>= 1) { if (tid < off) sm[tid] += sm[tid + off]; __syncthreads(); }
  const float Zs4 = sm[0]; __syncthreads();
  sm[tid] = et4; __syncthreads();
  for (int off = 64; off; off >>= 1) { if (tid < off) sm[tid] += sm[tid + off]; __syncthreads(); }
  const float Zt4 = sm[0]; __syncthreads();

  float klt = 0.f;
  if (ok) {
    const float tlp = (t - mt) * 0.25f - logf(Zt4);
    const float tp  = et4 / Zt4;
    const float slp = (s - ms) * 0.25f - logf(Zs4);
    klt = tp * (tlp - slp);
  }
  sm[tid] = klt; __syncthreads();
  for (int off = 64; off; off >>= 1) { if (tid < off) sm[tid] += sm[tid + off]; __syncthreads(); }
  if (tid == 0) {
    kl_part[row] = sm[0];
    const int lab = targets[row * 8];  // targets.reshape(B,-1)[:,0]
    ce_part[row]  = -(sl[row * NCLS + lab] - ms - logf(Zs));
  }
}

// ---------------------------------------------------------------------------
// Kernel 3: fused 1024x1024 outer-sum reduction (never materializes Ls/Lt).
// Six scalar sums:
//  S1 = sum (sI-tI)^2      S2 = sum (sC-tC)^2        S3 = sum_k (sP-tP)^2
//  S4 = sum_diag sI        S5 = sum_{cls eq} sC       S6 = sum_k (sP - gP)^2
// ---------------------------------------------------------------------------
__global__ void __launch_bounds__(256)
policy_reduce_kernel(const float* __restrict__ mats,
                     const float* __restrict__ bias1,
                     const float* __restrict__ bias2,
                     const int* __restrict__ targets,
                     float* __restrict__ pol_part) {
  __shared__ float red[256];
  const int tid = threadIdx.x;
  const unsigned idx = blockIdx.x * 256u + tid;
  const int i = idx >> 10;      // odd-row index (B half, t2)
  const int j = idx & 1023;     // even-row index (A half, t1)
  const float* ar  = mats + 0 * B1 * MATS + j * MATS;  // A_s
  const float* br  = mats + 1 * B1 * MATS + i * MATS;  // B_s
  const float* atr = mats + 2 * B1 * MATS + j * MATS;  // A_t
  const float* btr = mats + 3 * B1 * MATS + i * MATS;  // B_t
  const int* t1 = targets + 16 * j;      // row 2j
  const int* t2 = targets + 16 * i + 8;  // row 2i+1

  float acc[6];
  float sI = br[0] + ar[0] + bias2[0];
  float tI = btr[0] + atr[0] + bias1[0];
  float d  = sI - tI;
  acc[0] = d * d;
  acc[3] = (i == j) ? sI : 0.f;
  float sC = br[1] + ar[1] + bias2[1];
  float tC = btr[1] + atr[1] + bias1[1];
  d = sC - tC;
  acc[1] = d * d;
  acc[4] = (t1[0] == t2[0]) ? sC : 0.f;
  float s3 = 0.f, s6 = 0.f;
#pragma unroll
  for (int k = 0; k < 7; ++k) {
    const float sP = br[2 + k] + ar[2 + k] + bias2[2 + k];
    const float tP = btr[2 + k] + atr[2 + k] + bias1[2 + k];
    d = sP - tP; s3 += d * d;
    const float g = (t1[1 + k] == t2[1 + k]) ? 1.f : -1.f;
    d = sP - g;  s6 += d * d;
  }
  acc[2] = s3; acc[5] = s6;

#pragma unroll
  for (int c = 0; c < 6; ++c) {
    red[tid] = acc[c]; __syncthreads();
    for (int off = 128; off; off >>= 1) { if (tid < off) red[tid] += red[tid + off]; __syncthreads(); }
    if (tid == 0) pol_part[blockIdx.x * 6 + c] = red[0];
    __syncthreads();
  }
}

// ---------------------------------------------------------------------------
// Kernel 4: deterministic final combine in double precision.
// ---------------------------------------------------------------------------
__global__ void __launch_bounds__(256)
final_kernel(const float* __restrict__ ce_part, const float* __restrict__ kl_part,
             const float* __restrict__ pol_part, float* __restrict__ out) {
  __shared__ double red[256];
  const int tid = threadIdx.x;
  double a[8] = {0, 0, 0, 0, 0, 0, 0, 0};
  for (int r = tid; r < BTOT; r += 256) { a[0] += ce_part[r]; a[1] += kl_part[r]; }
  for (int b = tid; b < 4096; b += 256) {
#pragma unroll
    for (int c = 0; c < 6; ++c) a[2 + c] += pol_part[b * 6 + c];
  }
  double sums[8];
  for (int q = 0; q < 8; ++q) {
    red[tid] = a[q]; __syncthreads();
    for (int off = 128; off; off >>= 1) { if (tid < off) red[tid] += red[tid + off]; __syncthreads(); }
    sums[q] = red[0]; __syncthreads();
  }
  if (tid == 0) {
    const double ce  = sums[0] / 2048.0;
    const double kl  = sums[1] * 16.0 / (2048.0 * 100.0);  // KL_TEMP^2 / (B*C)
    const double inv = 1.0 / (1024.0 * 1024.0);
    const double S1 = sums[2], S2 = sums[3], S3 = sums[4];
    const double S4 = sums[5], S5 = sums[6], S6 = sums[7];
    // POS_W==NEG_W -> _wmean(loss,.,w,w) = w*mean(loss)
    const double kl_pol = S1 * inv + 0.5 * S2 * inv + 0.001 * S3 * inv / 7.0;
    const double gid = 1024.0 / 1023.0;  // identity_value
    const double kld_I = (1024.0 * gid * log(gid) - gid * S4) * inv; // diag-only KL
    const double ce_pol = kld_I - 0.5 * S5 * inv + 0.001 * S6 * inv / 7.0;
    out[0] = (float)(ce + kl + kl_pol + ce_pol);
  }
}

extern "C" void kernel_launch(void* const* d_in, const int* in_sizes, int n_in,
                              void* d_out, int out_size, void* d_ws, size_t ws_size,
                              hipStream_t stream) {
  (void)in_sizes; (void)n_in; (void)out_size; (void)ws_size;
  const float* sl = (const float*)d_in[0];
  const float* tl = (const float*)d_in[1];
  const float* sp = (const float*)d_in[2];
  const float* tp = (const float*)d_in[3];
  const float* W1 = (const float*)d_in[4];
  const float* b1 = (const float*)d_in[5];
  const float* W2 = (const float*)d_in[6];
  const float* b2 = (const float*)d_in[7];
  const int*  tgt = (const int*)d_in[8];

  float* ws       = (float*)d_ws;
  float* mats     = ws;                 // 4 * 1024 * 16 = 65536 floats
  float* ce_part  = ws + 65536;         // 2048
  float* kl_part  = ws + 67584;         // 2048
  float* pol_part = ws + 69632;         // 4096 * 6 = 24576

  pair_linear_wmma_kernel<<<64, 128, 0, stream>>>(sp, tp, W1, W2, mats);
  logits_kernel<<<BTOT, 128, 0, stream>>>(sl, tl, tgt, ce_part, kl_part);
  policy_reduce_kernel<<<4096, 256, 0, stream>>>(mats, b1, b2, tgt, pol_part);
  final_kernel<<<1, 256, 0, stream>>>(ce_part, kl_part, pol_part, (float*)d_out);
}
